// DeformableTransformer_48163763258131
// MI455X (gfx1250) — matrix operand order repaired and checked by reading, not compiled, not run on hardware
//
#include <hip/hip_runtime.h>

#define NB   8
#define LQ   900
#define DD   256
#define NHD  8
#define NLV  4
#define NPT  4
#define CC   32
#define SS   21760   // 128*128 + 64*64 + 32*32 + 16*16

typedef __attribute__((ext_vector_type(16))) __bf16 v16bf;
typedef __attribute__((ext_vector_type(8)))  float  v8f;

struct Frag { unsigned int d[8]; };

// Native f32 -> bf16 (RNE) conversions: let clang emit v_cvt*bf16* ops instead
// of a manual round-to-nearest-even integer sequence.
__device__ __forceinline__ unsigned short f2bf_u16(float f) {
    return __builtin_bit_cast(unsigned short, (__bf16)f);
}
__device__ __forceinline__ unsigned int cvt2bf(float f0, float f1) {
    unsigned short a = __builtin_bit_cast(unsigned short, (__bf16)f0);
    unsigned short b = __builtin_bit_cast(unsigned short, (__bf16)f1);
    return (unsigned int)a | ((unsigned int)b << 16);
}
__device__ __forceinline__ float bf2f(unsigned short h) {
    return __uint_as_float(((unsigned int)h) << 16);
}

__device__ __forceinline__ v8f wmma_bf16(const Frag& a, const Frag& b, v8f c) {
    v16bf av = __builtin_bit_cast(v16bf, a);
    v16bf bv = __builtin_bit_cast(v16bf, b);
    return __builtin_amdgcn_wmma_f32_16x16x32_bf16(false, av, false, bv,
                                                   (short)0, c, false, false);
}

// Build A hi/lo fragments for one 16x32 bf16 tile from row-major fp32.
// A layout (ISA 7.12.2): lanes 0-15 row m=lane, K={0..7,16..23}; lanes 16-31 same row, K={8..15,24..31}.
// `arow` already points at A[m][kt*32 + (lane<16 ? 0 : 8)].
__device__ __forceinline__ void build_a(const float* __restrict__ arow, Frag& hi, Frag& lo) {
    float4 q0 = *(const float4*)(arow + 0);
    float4 q1 = *(const float4*)(arow + 4);
    float4 q2 = *(const float4*)(arow + 16);
    float4 q3 = *(const float4*)(arow + 20);
    float fa[16] = {q0.x, q0.y, q0.z, q0.w, q1.x, q1.y, q1.z, q1.w,
                    q2.x, q2.y, q2.z, q2.w, q3.x, q3.y, q3.z, q3.w};
#pragma unroll
    for (int d = 0; d < 8; ++d) {
        float f0 = fa[2 * d], f1 = fa[2 * d + 1];
        unsigned int hp = cvt2bf(f0, f1);
        hi.d[d] = hp;
        // unpack the packed hi dword directly: low half -> <<16, high half -> mask
        float g0 = __uint_as_float(hp << 16);
        float g1 = __uint_as_float(hp & 0xffff0000u);
        lo.d[d] = cvt2bf(f0 - g0, f1 - g1);
    }
}

// Pack fp32 weights (row-major KxN, K=256, optionally concat of two matrices along N)
// into per-lane WMMA B fragments, hi+lo bf16 split.
// P layout: [frag f = kt*(N/16)+nt][lane 0..31][dword 0..15], dwords 0..7 = hi, 8..15 = lo.
// B layout (ISA 7.12.2): lane n = nt*16 + (lane&15); lanes<16: K={0..7,16..23}; lanes>=16: K={8..15,24..31}.
__global__ void pack_w_kernel(const float* __restrict__ W0, const float* __restrict__ W1,
                              int N0, int N1, unsigned int* __restrict__ P) {
    const int N = N0 + N1;
    int gid = blockIdx.x * blockDim.x + threadIdx.x;
    if (gid >= 256 * N) return;
    int f    = gid >> 9;
    int r    = gid & 511;
    int lane = r >> 4;
    int d16  = r & 15;
    int isLo = d16 >> 3;
    int d    = d16 & 7;
    int nfr  = N >> 4;
    int kt   = f / nfr, nt = f % nfr;
    int nn   = nt * 16 + (lane & 15);
    int k    = kt * 32 + 2 * d + ((d & 4) ? 8 : 0) + ((lane & 16) ? 8 : 0);
    float w0, w1;
    if (nn < N0) { w0 = W0[(size_t)k * N0 + nn];        w1 = W0[(size_t)(k + 1) * N0 + nn]; }
    else         { w0 = W1[(size_t)k * N1 + (nn - N0)]; w1 = W1[(size_t)(k + 1) * N1 + (nn - N0)]; }
    unsigned int hp = cvt2bf(w0, w1);
    unsigned int out;
    if (!isLo) {
        out = hp;
    } else {
        float g0 = __uint_as_float(hp << 16);
        float g1 = __uint_as_float(hp & 0xffff0000u);
        out = cvt2bf(w0 - g0, w1 - g1);
    }
    P[gid] = out;
}

__global__ void concat_bias_kernel(const float* __restrict__ b0, const float* __restrict__ b1,
                                   float* __restrict__ out) {
    int t = blockIdx.x * blockDim.x + threadIdx.x;
    if (t < 256) out[t] = b0[t];
    else if (t < 384) out[t] = b1[t - 256];
}

// D[MxN] = A[Mx256 fp32] @ Wpacked + bias, split-bf16 (3 WMMAs / fragment pair).
// One wave computes a 16x128 tile (8 column fragments). Block = 8 waves = 128 rows.
template <bool BF16_OUT>
__global__ __launch_bounds__(256) void gemm_k256(const float* __restrict__ A,
                                                 const unsigned int* __restrict__ Bp,
                                                 const float* __restrict__ bias,
                                                 void* __restrict__ Dp, int M, int N) {
    const int wave = threadIdx.x >> 5;
    const int lane = threadIdx.x & 31;
    const int m0 = (blockIdx.x * 8 + wave) * 16;
    if (m0 >= M) return;   // wave-uniform: EXEC stays all-ones for WMMA
    const int nfr_total = N >> 4;
    const int nt0 = blockIdx.y * 8;
    const int mrow = m0 + (lane & 15);
    const float* abase = A + (size_t)mrow * 256 + ((lane & 16) ? 8 : 0);

    v8f acc[8];
#pragma unroll
    for (int j = 0; j < 8; ++j) acc[j] = v8f{0.f, 0.f, 0.f, 0.f, 0.f, 0.f, 0.f, 0.f};

#pragma unroll
    for (int kt = 0; kt < 8; ++kt) {
        if (kt < 7) __builtin_prefetch(abase + (kt + 1) * 32, 0, 3);
        Frag ah, al;
        build_a(abase + kt * 32, ah, al);
#pragma unroll
        for (int j = 0; j < 8; ++j) {
            const unsigned int* bp =
                Bp + (((size_t)(kt * nfr_total + nt0 + j)) << 9) + (lane << 4);
            Frag bh, bl;
            *(uint4*)&bh.d[0] = *(const uint4*)(bp + 0);
            *(uint4*)&bh.d[4] = *(const uint4*)(bp + 4);
            *(uint4*)&bl.d[0] = *(const uint4*)(bp + 8);
            *(uint4*)&bl.d[4] = *(const uint4*)(bp + 12);
            acc[j] = wmma_bf16(ah, bh, acc[j]);   // hi*hi
            acc[j] = wmma_bf16(al, bh, acc[j]);   // lo*hi
            acc[j] = wmma_bf16(ah, bl, acc[j]);   // hi*lo
        }
    }

    // C/D layout: lane n = (lane&15), row = v + (lane>=16 ? 8 : 0)
    const int mb = m0 + ((lane & 16) ? 8 : 0);
#pragma unroll
    for (int j = 0; j < 8; ++j) {
        int n = (nt0 + j) * 16 + (lane & 15);
        float bv = bias[n];
#pragma unroll
        for (int v = 0; v < 8; ++v) {
            float val = acc[j][v] + bv;
            size_t di = (size_t)(mb + v) * N + n;
            if (BF16_OUT) ((unsigned short*)Dp)[di] = f2bf_u16(val);
            else          ((float*)Dp)[di] = val;
        }
    }
}

// One wave32 per (n, q, h). lane = channel c (0..31): every bilinear tap is one
// coalesced 64B bf16 request. Softmax(16) via wave shuffles.
__global__ __launch_bounds__(256) void msda_sample_kernel(
    const float* __restrict__ qry,          // (7200, 384): [0:256]=offsets, [256:384]=logits
    const float* __restrict__ refp,         // (NB, LQ, NL, 2)
    const unsigned short* __restrict__ val, // (NB, S, NH, 32) bf16
    float* __restrict__ sout)               // (7200, 256)
{
    const int wid  = blockIdx.x * 8 + (threadIdx.x >> 5);   // 0..57599
    const int lane = threadIdx.x & 31;
    const int h = wid & 7;
    const int q = (wid >> 3) % LQ;
    const int n = wid / (LQ * NHD);
    const size_t qrow = (size_t)n * LQ + q;
    const float* qp = qry + qrow * 384;

    // softmax over the 16 attention logits of this head
    float lg = (lane < 16) ? qp[256 + h * 16 + lane] : -3.0e38f;
    float mx = lg;
#pragma unroll
    for (int o = 16; o > 0; o >>= 1) mx = fmaxf(mx, __shfl_xor(mx, o, 32));
    float e = (lane < 16) ? __expf(lg - mx) : 0.f;
    float sm = e;
#pragma unroll
    for (int o = 16; o > 0; o >>= 1) sm += __shfl_xor(sm, o, 32);
    float w = e / sm;

    const int HW[4]  = {128, 64, 32, 16};
    const int LST[4] = {0, 16384, 20480, 21504};

    float acc = 0.f;
    const float* rp = refp + qrow * NLV * 2;
#pragma unroll
    for (int lvl = 0; lvl < NLV; ++lvl) {
        const int Hh = HW[lvl], Ww = HW[lvl];
        const float rx = rp[lvl * 2 + 0], ry = rp[lvl * 2 + 1];
        const unsigned short* vbase =
            val + (((size_t)n * SS + LST[lvl]) * NHD + h) * CC + lane;
        auto tap = [&](int yy, int xx) -> float {
            if (yy < 0 || yy >= Hh || xx < 0 || xx >= Ww) return 0.f;
            return bf2f(vbase[((size_t)yy * Ww + xx) * (NHD * CC)]);
        };
#pragma unroll
        for (int p = 0; p < NPT; ++p) {
            float ox = qp[h * 32 + lvl * 8 + p * 2 + 0];
            float oy = qp[h * 32 + lvl * 8 + p * 2 + 1];
            float x = rx * Ww + ox - 0.5f;   // (ref + off/W)*W - 0.5
            float y = ry * Hh + oy - 0.5f;
            float xf = floorf(x), yf = floorf(y);
            int ix = (int)xf, iy = (int)yf;
            float lx = x - xf, ly = y - yf;
            float aw = __shfl(w, lvl * 4 + p, 32);
            float v00 = tap(iy, ix);
            float v01 = tap(iy, ix + 1);
            float v10 = tap(iy + 1, ix);
            float v11 = tap(iy + 1, ix + 1);
            acc += aw * ((1.f - lx) * (1.f - ly) * v00 + lx * (1.f - ly) * v01 +
                         (1.f - lx) * ly * v10 + lx * ly * v11);
        }
    }
    sout[qrow * DD + h * CC + lane] = acc;
}

extern "C" void kernel_launch(void* const* d_in, const int* in_sizes, int n_in,
                              void* d_out, int out_size, void* d_ws, size_t ws_size,
                              hipStream_t stream) {
    const float* query  = (const float*)d_in[0];
    const float* refp   = (const float*)d_in[1];
    const float* inpf   = (const float*)d_in[2];
    // d_in[3] spatial_shapes, d_in[4] level_start_index: compile-time constants here
    const float* W_val  = (const float*)d_in[5];
    const float* b_val  = (const float*)d_in[6];
    const float* W_off  = (const float*)d_in[7];
    const float* b_off  = (const float*)d_in[8];
    const float* W_attn = (const float*)d_in[9];
    const float* b_attn = (const float*)d_in[10];
    const float* W_out  = (const float*)d_in[11];
    const float* b_out  = (const float*)d_in[12];

    char* ws = (char*)d_ws;
    size_t cur = 0;
    auto walloc = [&](size_t bytes) -> void* {
        void* p = ws + cur;
        cur += (bytes + 255) & ~(size_t)255;
        return p;
    };
    unsigned int* P_val = (unsigned int*)walloc((size_t)256 * 256 * 4);
    unsigned int* P_qry = (unsigned int*)walloc((size_t)256 * 384 * 4);
    unsigned int* P_out = (unsigned int*)walloc((size_t)256 * 256 * 4);
    float* b_qry        = (float*)walloc(384 * 4);
    float* qry_ws       = (float*)walloc((size_t)NB * LQ * 384 * 4);
    float* sout_ws      = (float*)walloc((size_t)NB * LQ * DD * 4);
    unsigned short* val_ws = (unsigned short*)walloc((size_t)NB * SS * DD * 2);

    // 1. Pack weights into WMMA B fragments (bf16 hi/lo split)
    pack_w_kernel<<<(256 * 256 + 255) / 256, 256, 0, stream>>>(W_val, W_val, 256, 0, P_val);
    pack_w_kernel<<<(256 * 384 + 255) / 256, 256, 0, stream>>>(W_off, W_attn, 256, 128, P_qry);
    pack_w_kernel<<<(256 * 256 + 255) / 256, 256, 0, stream>>>(W_out, W_out, 256, 0, P_out);
    concat_bias_kernel<<<2, 256, 0, stream>>>(b_off, b_attn, b_qry);

    // 2. value = input_flatten @ W_val + b_val  (M=174080), stored bf16
    {
        dim3 g(174080 / 128, 2);
        gemm_k256<true><<<g, 256, 0, stream>>>(inpf, P_val, b_val, val_ws, 174080, 256);
    }
    // 3. [off | attn_logits] = query @ [W_off | W_attn] + bias  (M=7200, N=384)
    {
        dim3 g((7200 + 127) / 128, 3);
        gemm_k256<false><<<g, 256, 0, stream>>>(query, P_qry, b_qry, qry_ws, 7200, 384);
    }
    // 4. softmax + deformable bilinear sampling → per-head outputs (7200, 256)
    msda_sample_kernel<<<(NB * LQ * NHD) / 8, 256, 0, stream>>>(qry_ws, refp, val_ws, sout_ws);
    // 5. out = sampled @ W_out + b_out → d_out
    {
        dim3 g((7200 + 127) / 128, 2);
        gemm_k256<false><<<g, 256, 0, stream>>>(sout_ws, P_out, b_out, (float*)d_out, 7200, 256);
    }
}
